// FieldWeightedFactorizationMachineModel_74783970558604
// MI455X (gfx1250) — compile-verified, matrix operand order checked by reading, not run on hardware
//
#include <hip/hip_runtime.h>
#include <hip/hip_bf16.h>
#include <math.h>

typedef __attribute__((ext_vector_type(16))) _Float16 v16h;
typedef __attribute__((ext_vector_type(8)))  float    v8f;

// Problem constants (from reference setup_inputs)
#define NF   39        // fields
#define ND   10        // embedding dim
#define NV   100000    // vocab per field
#define FP   48        // padded M dim (3x16)
#define KP   64        // padded K dim (2x32)
#define KP2  72        // padded f16 row length (144B rows: 16B aligned, bank-spread)
#define NBAT 16        // batch elements per block
#define NCOL (NBAT*ND) // 160 columns per block
#define NTHR 320       // 10 wave32

__global__ __launch_bounds__(NTHR)
void fwfm_fused_kernel(const int* __restrict__ x,          // (B, F) indices
                       const float* __restrict__ emb,      // (F, V, D)
                       const float* __restrict__ W,        // (F, F) cov
                       const float* __restrict__ Lw,       // (F, D) linear
                       float* __restrict__ out,            // (B,)
                       int Btotal) {
    __shared__ _Float16 s_cov[FP][KP2];    // symmetrized cov, f16, zero padded
    __shared__ _Float16 s_XhT[NCOL][KP2];  // transposed f16 X: [column][k]
    __shared__ float    s_X[FP][NCOL];     // fp32 X for epilogue dots (rows>=39 zero)
    __shared__ float    s_pair[NCOL];
    __shared__ float    s_first[NCOL];
    __shared__ float    s_diag[NCOL];

    const int tid   = threadIdx.x;
    const int bbase = blockIdx.x * NBAT;

    // ---- Phase 1: gather embeddings into LDS (the real cost of this kernel) ----
    // Convert to f16 ONCE here, stored transposed so B fragments are contiguous.
    for (int i = tid; i < KP * NCOL; i += NTHR) {
        int g = i / NCOL;
        int c = i % NCOL;
        float v = 0.0f;
        if (g < NF) {
            int b   = c / ND;
            int d   = c % ND;
            int row = x[(bbase + b) * NF + g];
            v = emb[((size_t)g * NV + (size_t)row) * ND + d];
        }
        s_XhT[c][g] = (_Float16)v;
        if (g < FP) s_X[g][c] = v;
    }
    // symmetrized covariance -> f16 A-matrix storage (row-major, padded)
    for (int i = tid; i < FP * KP; i += NTHR) {
        int f = i / KP;
        int g = i % KP;
        float v = 0.0f;
        if (f < NF && g < NF) v = 0.5f * (W[f * NF + g] + W[g * NF + f]);
        s_cov[f][g] = (_Float16)v;
    }
    if (tid < NCOL) s_pair[tid] = 0.0f;
    __syncthreads();

    // ---- Phase 2a: per-column first-order and diagonal terms (waves 0..4) ----
    if (tid < NCOL) {
        int c = tid;
        int d = c % ND;
        float fo = 0.0f, dg = 0.0f;
        for (int f = 0; f < NF; ++f) {
            float xv = s_X[f][c];
            fo += xv * Lw[f * ND + d];
            dg += W[f * NF + f] * xv * xv;   // diag(sym) == diag(W)
        }
        s_first[c] = fo;
        s_diag[c]  = dg;
    }

    // ---- Phase 2b: WMMA  M = S_sym @ X  then pair_c += sum_f X[f,c]*M[f,c] ----
    // All 10 waves execute with EXEC all-ones (WMMA requirement).
    {
        const int wave  = tid / 32;        // n-tile id, 0..9
        const int lane  = tid % 32;
        const int laneN = lane & 15;
        const int hi    = lane >> 4;       // 0: lanes 0-15, 1: lanes 16-31
        const int kofs  = hi * 8;          // A/B fragment K base per lane half
        const int c     = wave * 16 + laneN;
        const _Float16* __restrict__ bcol = &s_XhT[c][0];

        float pairpart = 0.0f;
        for (int mt = 0; mt < 3; ++mt) {
            v8f acc = {};
            const _Float16* __restrict__ arow = &s_cov[mt * 16 + laneN][0];
            for (int kt = 0; kt < 2; ++kt) {
                const int k0 = kt * 32 + kofs;       // 8 contiguous halfs
                const int k1 = k0 + 16;              // 8 contiguous halfs
                v16h a, bm;
#pragma unroll
                for (int i = 0; i < 8; ++i) {
                    a[i]      = arow[k0 + i];
                    a[i + 8]  = arow[k1 + i];
                    bm[i]     = bcol[k0 + i];
                    bm[i + 8] = bcol[k1 + i];
                }
                acc = __builtin_amdgcn_wmma_f32_16x16x32_f16(
                        /*neg_a=*/false, a, /*neg_b=*/false, bm,
                        /*c_mod=*/(short)0, acc,
                        /*reuse_a=*/false, /*reuse_b=*/false);
            }
            // D layout: lane holds (f = mt*16 + r + hi*8, n = laneN) for r=0..7.
            float p = 0.0f;
#pragma unroll
            for (int r = 0; r < 8; ++r) {
                int f = mt * 16 + r + hi * 8;
                p += acc[r] * s_X[f][c];   // rows f>=39 have X==0 -> no padding error
            }
            pairpart += p;
        }
        atomicAdd(&s_pair[c], pairpart);   // lane n and lane n+16 both hit column c
    }
    __syncthreads();

    // ---- Phase 3: per-batch reduction over D columns, sigmoid, store ----
    if (tid < NBAT) {
        int b = tid;
        float pair = 0.0f, fo = 0.0f, dg = 0.0f;
        for (int d = 0; d < ND; ++d) {
            int c = b * ND + d;
            pair += s_pair[c];
            fo   += s_first[c];
            dg   += s_diag[c];
        }
        float z = fo + 0.5f * (pair - dg);
        int bout = bbase + b;
        if (bout < Btotal) out[bout] = 1.0f / (1.0f + __expf(-z));
    }
}

extern "C" void kernel_launch(void* const* d_in, const int* in_sizes, int n_in,
                              void* d_out, int out_size, void* d_ws, size_t ws_size,
                              hipStream_t stream) {
    const int*   x   = (const int*)d_in[0];    // (B, F) int
    const float* emb = (const float*)d_in[1];  // (F, V, D) fp32
    const float* W   = (const float*)d_in[2];  // (F, F) fp32
    const float* Lw  = (const float*)d_in[3];  // (F, D) fp32
    float* out = (float*)d_out;

    int Btotal = in_sizes[0] / NF;             // 16384
    int blocks = (Btotal + NBAT - 1) / NBAT;   // 1024
    fwfm_fused_kernel<<<blocks, NTHR, 0, stream>>>(x, emb, W, Lw, out, Btotal);
}
